// Attention_146028888650
// MI455X (gfx1250) — matrix operand order repaired and checked by reading, not compiled, not run on hardware
//
#include <hip/hip_runtime.h>
#include <hip/hip_bf16.h>

typedef __attribute__((ext_vector_type(16))) _Float16 v16h;
typedef __attribute__((ext_vector_type(8)))  _Float16 v8h;
typedef __attribute__((ext_vector_type(8)))  float    v8f;
typedef __attribute__((ext_vector_type(4))) unsigned int u32x4;
typedef __attribute__((ext_vector_type(8))) int          i32x8;
typedef __attribute__((ext_vector_type(4))) int          i32x4;

#if __has_builtin(__builtin_amdgcn_tensor_load_to_lds)
#define HAVE_TDM 1
#else
#define HAVE_TDM 0
#endif

namespace {
constexpr int kDim   = 768;
constexpr int kHeads = 12;
constexpr int kHD    = 64;
constexpr int kB     = 2;
constexpr int kN     = 2048;
constexpr int kRows  = kB * kN;       // 4096
constexpr float kScale = 0.125f;      // 64^-0.5

// workspace offsets, in _Float16 elements (all 16B aligned)
constexpr size_t OFF_X16    = 0;                                     // [4096][768]
constexpr size_t OFF_WQT    = OFF_X16    + (size_t)kRows*kDim;       // [768][768]  (W^T: [n][k])
constexpr size_t OFF_WKVT   = OFF_WQT    + (size_t)kDim*kDim;        // [128][768]
constexpr size_t OFF_WPROJT = OFF_WKVT   + (size_t)128*kDim;         // [768][768]
constexpr size_t OFF_Q16    = OFF_WPROJT + (size_t)kDim*kDim;        // [B*H][N][64]
constexpr size_t OFF_K16    = OFF_Q16    + (size_t)kB*kHeads*kN*kHD; // [B][N][64]
constexpr size_t OFF_VT16   = OFF_K16    + (size_t)kB*kN*kHD;        // [B][64][N]
constexpr size_t OFF_O16    = OFF_VT16   + (size_t)kB*kN*kHD;        // [B*N][768]
} // namespace

// ---------------- fragment loaders (wave32 WMMA layouts) ----------------
// A-matrix 16x32 f16: lane m = lane&15; lane-half selects k offset 0/8;
// elements hold k in {koff..koff+7} then {koff+16..koff+23}  -> two 16B loads.
static __device__ inline v16h load_fragA(const _Float16* base, int ld, int row0, int k0) {
  const int lane = threadIdx.x & 31;
  const _Float16* p = base + (size_t)(row0 + (lane & 15)) * ld + k0 + ((lane >> 4) << 3);
  v8h lo = *(const v8h*)(p);
  v8h hi = *(const v8h*)(p + 16);
  v16h f;
#pragma unroll
  for (int i = 0; i < 8; ++i) { f[i] = lo[i]; f[i + 8] = hi[i]; }
  return f;
}

// B-matrix 32x16 f16 loaded from B^T storage BT[n][k] (row-major, ld = K):
// lane n = lane&15; lane-half selects k offset 0/8; elements hold k in {koff..koff+7} then {koff+16..koff+23}.
static __device__ inline v16h load_fragB(const _Float16* baseT, int ld, int n0, int k0) {
  const int lane = threadIdx.x & 31;
  const _Float16* p = baseT + (size_t)(n0 + (lane & 15)) * ld + k0 + ((lane >> 4) << 3);
  v8h lo = *(const v8h*)(p);
  v8h hi = *(const v8h*)(p + 16);
  v16h f;
#pragma unroll
  for (int i = 0; i < 8; ++i) { f[i] = lo[i]; f[i + 8] = hi[i]; }
  return f;
}

#define WMMA_F16(a, b, c) \
  __builtin_amdgcn_wmma_f32_16x16x32_f16(false, (a), false, (b), (short)0, (c), false, false)

// cross-lane reductions over the 16 lanes sharing a C-fragment row group
static __device__ inline float rmax16(float v) {
#pragma unroll
  for (int m = 1; m < 16; m <<= 1) v = fmaxf(v, __shfl_xor(v, m, 32));
  return v;
}
static __device__ inline float rsum16(float v) {
#pragma unroll
  for (int m = 1; m < 16; m <<= 1) v += __shfl_xor(v, m, 32);
  return v;
}

#if HAVE_TDM
// ---- Tensor Data Mover: 2D tile Global -> LDS (D# per CDNA5 ISA ch.8) ----
static __device__ inline void tdm_load_2d(unsigned lds_addr, const void* gaddr,
                                          unsigned tensor_d0, unsigned tensor_d1,
                                          unsigned tile_d0, unsigned tile_d1,
                                          unsigned stride0) {
  const unsigned long long ga = (unsigned long long)(uintptr_t)gaddr;
  u32x4 g0;
  g0[0] = 1u;                                              // count=1, user mode
  g0[1] = lds_addr;                                        // LDS byte address
  g0[2] = (unsigned)ga;                                    // global addr lo
  g0[3] = (unsigned)((ga >> 32) & 0x1FFFFFFu) | (2u << 30);// addr hi | type=2
  i32x8 g1;
  g1[0] = (int)(1u << 16);                                 // wg_mask=0, data_size=2B
  g1[1] = (int)(tensor_d0 << 16);                          // tensor_dim0[15:0]
  g1[2] = (int)(((tensor_d0 >> 16) & 0xFFFFu) | (tensor_d1 << 16));
  g1[3] = (int)(((tensor_d1 >> 16) & 0xFFFFu) | (tile_d0 << 16));
  g1[4] = (int)(tile_d1 & 0xFFFFu);                        // tile_dim1 (tile_dim2=0)
  g1[5] = (int)stride0;                                    // tensor_dim0_stride lo
  g1[6] = 0;
  g1[7] = 0;
  i32x4 z4 = {};
#if __has_include(<hip/amd_detail/amd_gfx1250_TDM.h>)
  i32x8 z8 = {};
  __builtin_amdgcn_tensor_load_to_lds(g0, g1, z4, z4, z8, 0);
#else
  __builtin_amdgcn_tensor_load_to_lds(g0, g1, z4, z4, 0);
#endif
}
#endif

// ---------------- kernel 1: fp32 -> f16 conversions (+weight transpose) ----------------
__global__ __launch_bounds__(256) void convert_all(const float* __restrict__ x,
                                                   const float* __restrict__ Wq,
                                                   const float* __restrict__ Wkv,
                                                   const float* __restrict__ Wproj,
                                                   _Float16* __restrict__ ws) {
  _Float16* x16  = ws + OFF_X16;
  _Float16* wqT  = ws + OFF_WQT;
  _Float16* wkvT = ws + OFF_WKVT;
  _Float16* wprT = ws + OFF_WPROJT;
  const int tid = blockIdx.x * blockDim.x + threadIdx.x;
  const int nth = gridDim.x * blockDim.x;
  for (int i = tid; i < kRows * kDim; i += nth) x16[i] = (_Float16)x[i];
  for (int i = tid; i < kDim * kDim; i += nth) {
    int n = i / kDim, k = i % kDim;
    wqT[i]  = (_Float16)Wq[(size_t)k * kDim + n];
    wprT[i] = (_Float16)Wproj[(size_t)k * kDim + n];
  }
  for (int i = tid; i < 128 * kDim; i += nth) {
    int n = i / kDim, k = i % kDim;
    wkvT[i] = (_Float16)Wkv[(size_t)k * 128 + n];
  }
}

// ---------------- kernel 2: fused QKV GEMM (32x64 tile per wave, 8 acc) ----------------
__global__ __launch_bounds__(256) void qkv_gemm(_Float16* __restrict__ ws) {
  const _Float16* x16  = ws + OFF_X16;
  const _Float16* wqT  = ws + OFF_WQT;
  const _Float16* wkvT = ws + OFF_WKVT;
  _Float16* q16  = ws + OFF_Q16;
  _Float16* k16  = ws + OFF_K16;
  _Float16* vt16 = ws + OFF_VT16;

  constexpr int kNT = (kDim + 2 * kHD) / 64;  // 14 n-tiles of 64
  const int tile = blockIdx.x * 8 + (threadIdx.x >> 5);
  if (tile >= (kRows / 32) * kNT) return;
  const int mt = tile / kNT, nt = tile % kNT;
  const int row0 = mt * 32, n0 = nt * 64;

  // hoist Wq-vs-Wkv base selection out of the k-loop (loop invariant)
  const _Float16* bb[4];
#pragma unroll
  for (int t = 0; t < 4; ++t) {
    const int n = n0 + t * 16;
    bb[t] = (n < kDim) ? wqT + (size_t)n * kDim
                       : wkvT + (size_t)(n - kDim) * kDim;
  }

  v8f acc0[4] = {{}, {}, {}, {}};
  v8f acc1[4] = {{}, {}, {}, {}};
#pragma unroll 2
  for (int k0 = 0; k0 < kDim; k0 += 32) {
    const v16h a0 = load_fragA(x16, kDim, row0, k0);       // reused across 4 WMMAs
    const v16h a1 = load_fragA(x16, kDim, row0 + 16, k0);  // reused across 4 WMMAs
#pragma unroll
    for (int t = 0; t < 4; ++t) {
      const v16h bfr = load_fragB(bb[t], kDim, 0, k0);     // reused across 2 WMMAs
      acc0[t] = WMMA_F16(a0, bfr, acc0[t]);
      acc1[t] = WMMA_F16(a1, bfr, acc1[t]);
    }
  }

  const int lane = threadIdx.x & 31;
  const int nloc = lane & 15, hi = lane >> 4;
#pragma unroll
  for (int g = 0; g < 2; ++g) {
    const v8f* acc = g ? acc1 : acc0;
#pragma unroll
    for (int t = 0; t < 4; ++t) {
#pragma unroll
      for (int v = 0; v < 8; ++v) {
        const int row = row0 + g * 16 + v + 8 * hi;
        const int b   = row / kN, i = row % kN;
        const int n   = n0 + t * 16 + nloc;
        const _Float16 val = (_Float16)acc[t][v];
        if (n < kDim) {                           // Q -> [B,H,N,64]
          q16[(((size_t)b * kHeads + (n >> 6)) * kN + i) * kHD + (n & 63)] = val;
        } else if (n < kDim + kHD) {              // K -> [B,N,64]
          k16[((size_t)b * kN + i) * kHD + (n - kDim)] = val;
        } else {                                  // V -> transposed [B,64,N]
          vt16[((size_t)b * kHD + (n - kDim - kHD)) * kN + i] = val;
        }
      }
    }
  }
}

// ---------------- kernel 3: flash-style multi-query attention ----------------
// Block = 8 waves of the SAME (b,h): K/V tiles are shared, so wave 0 streams
// them into LDS with double-buffered TDM tensor loads; all 8 waves consume.
__global__ __launch_bounds__(256) void attention(_Float16* __restrict__ ws) {
#if HAVE_TDM
  __shared__ _Float16 ktile[2][32 * 64];  // K rows j..j+31, 64 cols
  __shared__ _Float16 vtile[2][64 * 32];  // V^T rows d=0..63, 32 cols
#endif
  __shared__ _Float16 plds[8][16 * 32];   // per-wave P staging tile

  const _Float16* q16  = ws + OFF_Q16;
  const _Float16* k16  = ws + OFF_K16;
  const _Float16* vt16 = ws + OFF_VT16;
  _Float16* o16 = ws + OFF_O16;

  const int widx = threadIdx.x >> 5;
  const int bh = blockIdx.x >> 4;                 // 24 (b,h) groups x 16 blocks
  const int it = ((blockIdx.x & 15) << 3) + widx; // 128 query tiles per (b,h)
  const int h = bh % kHeads, b = bh / kHeads;
  const int i0 = it * 16;

  const _Float16* qbase  = q16 + (size_t)bh * kN * kHD;
  const _Float16* kbase  = k16 + (size_t)b * kN * kHD;
  const _Float16* vtbase = vt16 + (size_t)b * kHD * kN;
  _Float16* pl = plds[widx];

  const v16h qa0 = load_fragA(qbase, kHD, i0, 0);
  const v16h qa1 = load_fragA(qbase, kHD, i0, 32);

  float m[8], l[8];
  v8f acc[4] = {{}, {}, {}, {}};
#pragma unroll
  for (int v = 0; v < 8; ++v) { m[v] = -1e30f; l[v] = 0.f; }

  const int lane = threadIdx.x & 31;
  const int nloc = lane & 15, hi = lane >> 4;

#if HAVE_TDM
  if (widx == 0) {  // prologue: fill buffer 0
    tdm_load_2d((unsigned)(uintptr_t)&ktile[0][0], kbase, kHD, kN, kHD, 32, kHD);
    tdm_load_2d((unsigned)(uintptr_t)&vtile[0][0], vtbase, kN, kHD, 32, kHD, kN);
  }
#endif

  for (int jt = 0; jt < kN / 32; ++jt) {
    const int j0 = jt * 32;
#if HAVE_TDM
    if (widx == 0) {
      if (jt + 1 < kN / 32) {  // prefetch next j-tile into the other buffer
        const int jn = j0 + 32, nb = (jt + 1) & 1;
        tdm_load_2d((unsigned)(uintptr_t)&ktile[nb][0], kbase + (size_t)jn * kHD,
                    kHD, kN, kHD, 32, kHD);
        tdm_load_2d((unsigned)(uintptr_t)&vtile[nb][0], vtbase + jn,
                    kN, kHD, 32, kHD, kN);
        __builtin_amdgcn_s_wait_tensorcnt(2);  // current tile's 2 DMAs done
      } else {
        __builtin_amdgcn_s_wait_tensorcnt(0);
      }
    }
    __syncthreads();
    const _Float16* kt = ktile[jt & 1]; const int kld = kHD; const int krow = 0;
    const _Float16* vt = vtile[jt & 1]; const int vld = 32;  const int vcol = 0;
#else
    const _Float16* kt = kbase;  const int kld = kHD; const int krow = j0;
    const _Float16* vt = vtbase; const int vld = kN;  const int vcol = j0;
#endif

    // S = Q K^T
    v8f s0 = {}, s1 = {};
    s0 = WMMA_F16(qa0, load_fragB(kt, kld, krow, 0), s0);
    s0 = WMMA_F16(qa1, load_fragB(kt, kld, krow, 32), s0);
    s1 = WMMA_F16(qa0, load_fragB(kt, kld, krow + 16, 0), s1);
    s1 = WMMA_F16(qa1, load_fragB(kt, kld, krow + 16, 32), s1);

    float p0[8], p1[8], corr[8];
#pragma unroll
    for (int v = 0; v < 8; ++v) {
      const float a = s0[v] * kScale, c = s1[v] * kScale;
      const float tmax = rmax16(fmaxf(a, c));
      const float mn = fmaxf(m[v], tmax);
      corr[v] = __expf(m[v] - mn);
      m[v] = mn;
      p0[v] = __expf(a - mn);
      p1[v] = __expf(c - mn);
      l[v] = l[v] * corr[v] + rsum16(p0[v] + p1[v]);
    }
#pragma unroll
    for (int t = 0; t < 4; ++t)
#pragma unroll
      for (int v = 0; v < 8; ++v) acc[t][v] *= corr[v];

    // stage P (C layout, f32) -> LDS (row-major 16x32 f16) -> A fragment
#pragma unroll
    for (int v = 0; v < 8; ++v) {
      const int M = v + 8 * hi;
      pl[M * 32 + nloc]      = (_Float16)p0[v];
      pl[M * 32 + nloc + 16] = (_Float16)p1[v];
    }
    asm volatile("s_wait_dscnt 0" ::: "memory");
    const v16h pf = load_fragA(pl, 32, 0, 0);

    // acc += P @ V
#pragma unroll
    for (int t = 0; t < 4; ++t)
      acc[t] = WMMA_F16(pf, load_fragB(vt, vld, t * 16, vcol), acc[t]);

#if HAVE_TDM
    __syncthreads();  // all waves done with this buffer before it is refilled
#endif
  }

  // normalize and scatter to [B,N, H*64+d] (f16 input for the projection GEMM)
#pragma unroll
  for (int v = 0; v < 8; ++v) {
    const float inv = 1.f / l[v];
    const int i = i0 + v + 8 * hi;
#pragma unroll
    for (int t = 0; t < 4; ++t)
      o16[((size_t)b * kN + i) * kDim + h * kHD + t * 16 + nloc] = (_Float16)(acc[t][v] * inv);
  }
}

// ---------------- kernel 4: output projection + bias (32x64 per wave) ----------------
__global__ __launch_bounds__(256) void proj_gemm(const _Float16* __restrict__ ws,
                                                 const float* __restrict__ bproj,
                                                 float* __restrict__ out) {
  const _Float16* o16  = ws + OFF_O16;
  const _Float16* wprT = ws + OFF_WPROJT;

  constexpr int kNT = kDim / 64;  // 12
  const int tile = blockIdx.x * 8 + (threadIdx.x >> 5);
  if (tile >= (kRows / 32) * kNT) return;
  const int mt = tile / kNT, nt = tile % kNT;
  const int row0 = mt * 32, n0 = nt * 64;

  v8f acc0[4] = {{}, {}, {}, {}};
  v8f acc1[4] = {{}, {}, {}, {}};
#pragma unroll 2
  for (int k0 = 0; k0 < kDim; k0 += 32) {
    const v16h a0 = load_fragA(o16, kDim, row0, k0);
    const v16h a1 = load_fragA(o16, kDim, row0 + 16, k0);
#pragma unroll
    for (int t = 0; t < 4; ++t) {
      const v16h bfr = load_fragB(wprT, kDim, n0 + t * 16, k0);
      acc0[t] = WMMA_F16(a0, bfr, acc0[t]);
      acc1[t] = WMMA_F16(a1, bfr, acc1[t]);
    }
  }

  const int lane = threadIdx.x & 31;
  const int nloc = lane & 15, hi = lane >> 4;
#pragma unroll
  for (int t = 0; t < 4; ++t) {
    const int n = n0 + t * 16 + nloc;
    const float bias = bproj[n];
#pragma unroll
    for (int v = 0; v < 8; ++v) {
      const int r0 = row0 + v + 8 * hi;
      out[(size_t)r0 * kDim + n] = acc0[t][v] + bias;
      out[(size_t)(r0 + 16) * kDim + n] = acc1[t][v] + bias;
    }
  }
}

extern "C" void kernel_launch(void* const* d_in, const int* in_sizes, int n_in,
                              void* d_out, int out_size, void* d_ws, size_t ws_size,
                              hipStream_t stream) {
  const float* x     = (const float*)d_in[0];
  const float* Wq    = (const float*)d_in[1];
  const float* Wkv   = (const float*)d_in[2];
  const float* Wproj = (const float*)d_in[3];
  const float* bproj = (const float*)d_in[4];
  _Float16* ws = (_Float16*)d_ws;
  float* out = (float*)d_out;

  convert_all<<<1024, 256, 0, stream>>>(x, Wq, Wkv, Wproj, ws);
  qkv_gemm<<<(kRows / 32) * ((kDim + 2 * kHD) / 64) / 8, 256, 0, stream>>>(ws);
  attention<<<kB * kHeads * (kN / 16) / 8, 256, 0, stream>>>(ws);
  proj_gemm<<<(kRows / 32) * (kDim / 64) / 8, 256, 0, stream>>>(ws, bproj, out);
}
